// QL_MATCC_Model_86388972191855
// MI455X (gfx1250) — matrix operand order repaired and checked by reading, not compiled, not run on hardware
//
#include <hip/hip_runtime.h>
#include <math.h>

// ---------------- Model constants ----------------
#define Lc 4
#define Cc 256
#define Bc 256
#define Tc 512
#define DIN 8
#define MTOK (Bc * Tc)          // 131072 rows
#define FF (4 * Cc)             // 1024

typedef __attribute__((ext_vector_type(16))) __bf16 v16bf;
typedef __attribute__((ext_vector_type(8)))  __bf16 v8bf;
typedef __attribute__((ext_vector_type(8)))  float  v8f;

__device__ __forceinline__ float gelu_exact(float x) {
    return 0.5f * x * (1.0f + erff(x * 0.70710678118654752f));
}

// ---------------- CDNA5 async global->LDS support (probe via __has_builtin) ----
#if __has_builtin(__builtin_amdgcn_global_load_async_to_lds_b128)
#define HAVE_ASYNC_LDS 1
#endif

#if __has_builtin(__builtin_amdgcn_s_wait_asynccnt)
#define WAIT_ASYNC(n) __builtin_amdgcn_s_wait_asynccnt(n)
#else
#define WAIT_ASYNC(n) asm volatile("s_wait_asynccnt %0" ::"n"(n) : "memory")
#endif

#ifdef HAVE_ASYNC_LDS
// Builtin prototype (from hipcc diagnostic): arg0 = int4 addrspace(1)*,
// arg1 = int4 addrspace(3)*, then imm offset, imm cpol.
typedef int v4i __attribute__((vector_size(16)));
typedef v4i __attribute__((address_space(1)))* as1_v4i_p;
typedef v4i __attribute__((address_space(3)))* as3_v4i_p;

// 16B memory -> LDS DMA, tracked by ASYNCcnt.
__device__ __forceinline__ void async_cp16(const __bf16* g, __bf16* l) {
    __builtin_amdgcn_global_load_async_to_lds_b128((as1_v4i_p)(g), (as3_v4i_p)(l), 0, 0);
}
#endif

// ---------------- CMA decomposition + decoder matmuls ----------------
__global__ __launch_bounds__(256)
void dec_kernel(const float* __restrict__ x, const float* __restrict__ Wt,
                const float* __restrict__ bt, const float* __restrict__ Wf,
                const float* __restrict__ bfv, float* __restrict__ h) {
    int bt_idx = blockIdx.x;            // b*T + t
    int b = bt_idx >> 9;
    int t = bt_idx & (Tc - 1);
    int c = threadIdx.x;
    __shared__ float tr[DIN], fl[DIN];
    if (c < DIN) {
        int lo = (t - 4 < 0) ? 0 : (t - 4);
        float s = 0.f;
        for (int tt = lo; tt <= t; ++tt) s += x[((size_t)b * Tc + tt) * DIN + c];
        float trd = s / (float)(t - lo + 1);
        tr[c] = trd;
        fl[c] = x[((size_t)b * Tc + t) * DIN + c] - trd;
    }
    __syncthreads();
    float acc = bt[c] + bfv[c];
#pragma unroll
    for (int d = 0; d < DIN; ++d)
        acc += tr[d] * Wt[d * Cc + c] + fl[d] * Wf[d * Cc + c];
    h[(size_t)bt_idx * Cc + c] = acc;
}

// ---------------- conv1: (B,2,T)->(B,32,T), pad 1, k=3, + gelu ----------------
__global__ __launch_bounds__(256)
void conv1_kernel(const float* __restrict__ x, const float* __restrict__ W1,
                  const float* __restrict__ b1, float* __restrict__ g1) {
    int idx = blockIdx.x * 256 + threadIdx.x;
    if (idx >= Bc * 32 * Tc) return;
    int t = idx & (Tc - 1);
    int j = (idx >> 9) & 31;
    int b = idx >> 14;
    float acc = b1[j];
#pragma unroll
    for (int i = 0; i < 2; ++i)
#pragma unroll
        for (int r = 0; r < 3; ++r) {
            int tt = t + r - 1;
            if (tt >= 0 && tt < Tc)
                acc += W1[j * 6 + i * 3 + r] * x[((size_t)b * Tc + tt) * DIN + 5 + i];
        }
    g1[((size_t)b * 32 + j) * Tc + t] = gelu_exact(acc);
}

// ---------------- conv2: (B,32,T)->(B,C,T) ----------------
__global__ __launch_bounds__(256)
void conv2_kernel(const float* __restrict__ g1, const float* __restrict__ W2,
                  const float* __restrict__ b2, float* __restrict__ g2) {
    int idx = blockIdx.x * 256 + threadIdx.x;
    if (idx >= Bc * Cc * Tc) return;
    int t = idx & (Tc - 1);
    int c = (idx >> 9) & (Cc - 1);
    int b = idx >> 17;
    float acc = b2[c];
    const float* w = W2 + c * 96;
    const float* gi = g1 + (size_t)b * 32 * Tc;
#pragma unroll 4
    for (int j = 0; j < 32; ++j)
#pragma unroll
        for (int r = 0; r < 3; ++r) {
            int tt = t + r - 1;
            if (tt >= 0 && tt < Tc) acc += w[j * 3 + r] * gi[j * Tc + tt];
        }
    g2[((size_t)b * Cc + c) * Tc + t] = acc;
}

// ---------------- per-batch mean/rstd over (C,T) ----------------
__global__ __launch_bounds__(256)
void bstats_kernel(const float* __restrict__ g2, float* __restrict__ stats) {
    int b = blockIdx.x;
    int tid = threadIdx.x;
    float s = 0.f, s2 = 0.f;
    size_t base = (size_t)b * (Cc * Tc);
    for (int i = tid; i < Cc * Tc; i += 256) {
        float v = g2[base + i];
        s += v; s2 += v * v;
    }
    __shared__ float sh[256], sh2[256];
    sh[tid] = s; sh2[tid] = s2;
    __syncthreads();
    for (int off = 128; off > 0; off >>= 1) {
        if (tid < off) { sh[tid] += sh[tid + off]; sh2[tid] += sh2[tid + off]; }
        __syncthreads();
    }
    if (tid == 0) {
        float mu = sh[0] / (float)(Cc * Tc);
        float var = sh2[0] / (float)(Cc * Tc) - mu * mu;
        stats[b * 2] = mu;
        stats[b * 2 + 1] = rsqrtf(var + 1e-5f);
    }
}

// ---------------- h += norm(g2)*mg_g + mg_be  (transposed back) ----------------
__global__ __launch_bounds__(256)
void gadd_kernel(const float* __restrict__ g2, const float* __restrict__ stats,
                 const float* __restrict__ gg, const float* __restrict__ gb,
                 float* __restrict__ h) {
    int bt_idx = blockIdx.x;
    int b = bt_idx >> 9;
    int t = bt_idx & (Tc - 1);
    int c = threadIdx.x;
    float mu = stats[b * 2], rs = stats[b * 2 + 1];
    float gv = (g2[((size_t)b * Cc + c) * Tc + t] - mu) * rs * gg[c] + gb[c];
    h[(size_t)bt_idx * Cc + c] += gv;
}

// ---------------- LayerNorm over C=256; optional sum of two inputs; dual output ----------------
__global__ __launch_bounds__(256)
void ln_kernel(const float* __restrict__ in1, const float* __restrict__ in2,
               const float* __restrict__ g, const float* __restrict__ b,
               float* __restrict__ outF, __bf16* __restrict__ outH) {
    int row = blockIdx.x;
    int c = threadIdx.x;
    size_t base = (size_t)row * Cc + c;
    float v = in1[base];
    if (in2) v += in2[base];
    __shared__ float s[256];
    s[c] = v;
    __syncthreads();
    for (int off = 128; off > 0; off >>= 1) {
        if (c < off) s[c] += s[c + off];
        __syncthreads();
    }
    float mu = s[0] * (1.0f / Cc);
    __syncthreads();
    float d = v - mu;
    s[c] = d * d;
    __syncthreads();
    for (int off = 128; off > 0; off >>= 1) {
        if (c < off) s[c] += s[c + off];
        __syncthreads();
    }
    float y = d * rsqrtf(s[0] * (1.0f / Cc) + 1e-5f) * g[c] + b[c];
    if (outF) outF[base] = y;
    if (outH) outH[base] = (__bf16)y;
}

// ---------------- weight transpose + f32->bf16 convert: W(KxN) -> Wt(NxK) ----------------
__global__ __launch_bounds__(256)
void wconv_kernel(const float* __restrict__ W, __bf16* __restrict__ Wt, int K, int N) {
    int idx = blockIdx.x * 256 + threadIdx.x;
    if (idx >= K * N) return;
    int k = idx / N, n = idx % N;
    Wt[(size_t)n * K + k] = (__bf16)W[idx];
}

// ---------------- RWKV scan: serial in T, parallel over (b,c) ----------------
__global__ __launch_bounds__(256)
void rwkv_scan_kernel(const float* __restrict__ v, const float* __restrict__ rpre,
                      const float* __restrict__ decay, const float* __restrict__ first,
                      __bf16* __restrict__ rw) {
    int idx = blockIdx.x * 256 + threadIdx.x;   // b*C + c
    int b = idx >> 8;
    int c = idx & (Cc - 1);
    float td = decay[c], tf = first[c];
    float aa = 0.f, bb = 0.f, pp = -10000.f;
    size_t base = (size_t)b * Tc * Cc + c;
    for (int t = 0; t < Tc; ++t) {
        size_t o = base + (size_t)t * Cc;
        float vt = v[o];
        float ww = tf + pp;
        float p = fmaxf(pp, ww);
        float e1 = __expf(pp - p), e2 = __expf(ww - p);
        float wkv = (e1 * aa + e2 * vt) / (e1 * bb + e2 + 1e-6f);
        float r = 1.f / (1.f + __expf(-rpre[o]));
        rw[o] = (__bf16)(r * wkv);
        float ww2 = td + pp;
        float p2 = fmaxf(pp, ww2);
        float f1 = __expf(pp - p2), f2 = __expf(ww2 - p2);
        aa = f1 * aa + f2 * vt;
        bb = f1 * bb + f2;
        pp = p2;
    }
}

// ---------------- WMMA GEMM: out(MxN) = A(MxK,bf16) * Bt(NxK,bf16)^T ----------------
// Block tile 128x128, 8 waves (4x2), wave tile 32x64 (2x4 wmma accs), K-step 32.
// Async path: double-buffered LDS, tiles staged by GLOBAL_LOAD_ASYNC_TO_LDS_B128
// (ASYNCcnt-tracked DMA) so stage i+1's copies overlap stage i's WMMAs.
#define G_STORE 0
#define G_ADD 1
#define G_GELU_BF16 2

template <int MODE>
__global__ __launch_bounds__(256)
void gemm_wmma_kernel(const __bf16* __restrict__ A, const __bf16* __restrict__ Bt,
                      const float* __restrict__ bias, float* __restrict__ outF,
                      __bf16* __restrict__ outH, int M, int N, int K) {
#ifdef HAVE_ASYNC_LDS
    __shared__ __align__(16) __bf16 sA[2][128 * 40];
    __shared__ __align__(16) __bf16 sB[2][128 * 40];
#else
    __shared__ __align__(16) __bf16 sA[1][128 * 40];
    __shared__ __align__(16) __bf16 sB[1][128 * 40];
#endif
    int tid = threadIdx.x;
    int nb = N >> 7;
    int bm = blockIdx.x / nb, bn = blockIdx.x % nb;
    int m0 = bm << 7, n0 = bn << 7;
    int wave = tid >> 5, lane = tid & 31;
    int wm = (wave >> 1) << 5;   // 0,32,64,96
    int wn = (wave & 1) << 6;    // 0,64

    v8f acc[2][4];
#pragma unroll
    for (int mi = 0; mi < 2; ++mi)
#pragma unroll
        for (int ni = 0; ni < 4; ++ni)
#pragma unroll
            for (int r = 0; r < 8; ++r) acc[mi][ni][r] = 0.f;

    int ldRow = tid >> 1;
    int ldCol = (tid & 1) << 4;
    const __bf16* gA = A + (size_t)(m0 + ldRow) * K + ldCol;
    const __bf16* gB = Bt + (size_t)(n0 + ldRow) * K + ldCol;

    int khalfA = (lane >> 4) << 4;   // 0 / 16
    int kbaseB = (lane >> 4) << 4;   // 0 / 16
    int lrow = lane & 15;

#ifdef HAVE_ASYNC_LDS
    // ---------- async double-buffered pipeline ----------
    int nstages = K >> 5;
    __bf16* lA0 = &sA[0][ldRow * 40 + ldCol];
    __bf16* lA1 = &sA[1][ldRow * 40 + ldCol];
    __bf16* lB0 = &sB[0][ldRow * 40 + ldCol];
    __bf16* lB1 = &sB[1][ldRow * 40 + ldCol];

    // prologue: stages 0 and 1 in flight
    async_cp16(gA, lA0);      async_cp16(gA + 8, lA0 + 8);
    async_cp16(gB, lB0);      async_cp16(gB + 8, lB0 + 8);
    if (nstages > 1) {
        async_cp16(gA + 32, lA1); async_cp16(gA + 40, lA1 + 8);
        async_cp16(gB + 32, lB1); async_cp16(gB + 40, lB1 + 8);
    }

    for (int i = 0; i < nstages; ++i) {
        int p = i & 1;
        // waits until at most 8 ops remain; stages are 2+2 ops, so stage i may still be in flight
        if (i + 1 < nstages) { WAIT_ASYNC(8); } else { WAIT_ASYNC(0); }
        __syncthreads();

        const __bf16* tileA = sA[p];
        const __bf16* tileB = sB[p];
        v16bf af[2], bfr[4];
#pragma unroll
        for (int mi = 0; mi < 2; ++mi) {
            const __bf16* pa = &tileA[(wm + mi * 16 + lrow) * 40];
            union { v16bf v; v8bf h[2]; } u;
            u.h[0] = *(const v8bf*)(pa + khalfA);
            u.h[1] = *(const v8bf*)(pa + khalfA + 8);
            af[mi] = u.v;
        }
#pragma unroll
        for (int ni = 0; ni < 4; ++ni) {
            const __bf16* pb = &tileB[(wn + ni * 16 + lrow) * 40 + kbaseB];
            union { v16bf v; v8bf h[2]; } u;
            u.h[0] = *(const v8bf*)(pb);
            u.h[1] = *(const v8bf*)(pb + 8);
            bfr[ni] = u.v;
        }
#pragma unroll
        for (int mi = 0; mi < 2; ++mi)
#pragma unroll
            for (int ni = 0; ni < 4; ++ni)
                acc[mi][ni] = __builtin_amdgcn_wmma_f32_16x16x32_bf16(
                    false, af[mi], false, bfr[ni], (short)0, acc[mi][ni], false, false);

        if (i + 2 < nstages) {
            __syncthreads();   // everyone done reading buffer p before refilling it
            int k2 = (i + 2) << 5;
            __bf16* dA = p ? lA1 : lA0;
            __bf16* dB = p ? lB1 : lB0;
            async_cp16(gA + k2, dA);      async_cp16(gA + k2 + 8, dA + 8);
            async_cp16(gB + k2, dB);      async_cp16(gB + k2 + 8, dB + 8);
        }
    }
#else
    // ---------- synchronous fallback ----------
    for (int k0 = 0; k0 < K; k0 += 32) {
        uint4 a0 = *(const uint4*)(gA + k0);
        uint4 a1 = *(const uint4*)(gA + k0 + 8);
        uint4 b0 = *(const uint4*)(gB + k0);
        uint4 b1 = *(const uint4*)(gB + k0 + 8);
        __syncthreads();
        *(uint4*)&sA[0][ldRow * 40 + ldCol] = a0;
        *(uint4*)&sA[0][ldRow * 40 + ldCol + 8] = a1;
        *(uint4*)&sB[0][ldRow * 40 + ldCol] = b0;
        *(uint4*)&sB[0][ldRow * 40 + ldCol + 8] = b1;
        __syncthreads();

        v16bf af[2], bfr[4];
#pragma unroll
        for (int mi = 0; mi < 2; ++mi) {
            const __bf16* pa = &sA[0][(wm + mi * 16 + lrow) * 40];
            union { v16bf v; v8bf h[2]; } u;
            u.h[0] = *(const v8bf*)(pa + khalfA);
            u.h[1] = *(const v8bf*)(pa + khalfA + 8);
            af[mi] = u.v;
        }
#pragma unroll
        for (int ni = 0; ni < 4; ++ni) {
            const __bf16* pb = &sB[0][(wn + ni * 16 + lrow) * 40 + kbaseB];
            union { v16bf v; v8bf h[2]; } u;
            u.h[0] = *(const v8bf*)(pb);
            u.h[1] = *(const v8bf*)(pb + 8);
            bfr[ni] = u.v;
        }
#pragma unroll
        for (int mi = 0; mi < 2; ++mi)
#pragma unroll
            for (int ni = 0; ni < 4; ++ni)
                acc[mi][ni] = __builtin_amdgcn_wmma_f32_16x16x32_bf16(
                    false, af[mi], false, bfr[ni], (short)0, acc[mi][ni], false, false);
    }
#endif

    // Epilogue. C/D layout: elem r of v8f -> row (r + 8*(lane>>4)), col (lane&15).
    int mlo = (lane >> 4) << 3;
    int nn = lane & 15;
#pragma unroll
    for (int mi = 0; mi < 2; ++mi)
#pragma unroll
        for (int ni = 0; ni < 4; ++ni) {
            int nIdx = n0 + wn + ni * 16 + nn;
            float bv = (MODE != G_ADD && bias) ? bias[nIdx] : 0.f;
#pragma unroll
            for (int r = 0; r < 8; ++r) {
                int mIdx = m0 + wm + mi * 16 + mlo + r;
                size_t o = (size_t)mIdx * N + nIdx;
                float val = acc[mi][ni][r] + bv;
                if (MODE == G_STORE) outF[o] = val;
                if (MODE == G_ADD) outF[o] += acc[mi][ni][r];
                if (MODE == G_GELU_BF16) outH[o] = (__bf16)gelu_exact(val);
            }
        }
}

// ---------------- head: out[b] = h[b,T-1,:] @ head_W + head_b ----------------
__global__ __launch_bounds__(256)
void head_kernel(const float* __restrict__ h, const float* __restrict__ hw,
                 const float* __restrict__ hb, float* __restrict__ out) {
    int b = blockIdx.x;
    int c = threadIdx.x;
    __shared__ float s[256];
    s[c] = h[((size_t)b * Tc + (Tc - 1)) * Cc + c] * hw[c];
    __syncthreads();
    for (int off = 128; off > 0; off >>= 1) {
        if (c < off) s[c] += s[c + off];
        __syncthreads();
    }
    if (c == 0) out[b] = s[0] + hb[0];
}

// ---------------- launcher ----------------
extern "C" void kernel_launch(void* const* d_in, const int* in_sizes, int n_in,
                              void* d_out, int out_size, void* d_ws, size_t ws_size,
                              hipStream_t stream) {
    const float* x       = (const float*)d_in[0];
    // d_in[1] = vol (unused by reference)
    const float* dec_Wt  = (const float*)d_in[2];
    const float* dec_bt  = (const float*)d_in[3];
    const float* dec_Wf  = (const float*)d_in[4];
    const float* dec_bf  = (const float*)d_in[5];
    const float* mg_W1   = (const float*)d_in[6];
    const float* mg_b1   = (const float*)d_in[7];
    const float* mg_W2   = (const float*)d_in[8];
    const float* mg_b2   = (const float*)d_in[9];
    const float* mg_g    = (const float*)d_in[10];
    const float* mg_be   = (const float*)d_in[11];
    const float* tm_decay= (const float*)d_in[12];
    const float* tm_first= (const float*)d_in[13];
    // d_in[14] = tm_Wk (unused by reference)
    const float* tm_Wv   = (const float*)d_in[15];
    const float* tm_Wr   = (const float*)d_in[16];
    const float* tm_Wo   = (const float*)d_in[17];
    const float* ln1_g   = (const float*)d_in[18];
    const float* ln1_b   = (const float*)d_in[19];
    const float* ffn_W1  = (const float*)d_in[20];
    const float* ffn_b1  = (const float*)d_in[21];
    const float* ffn_W2  = (const float*)d_in[22];
    const float* ffn_b2  = (const float*)d_in[23];
    const float* cm_g    = (const float*)d_in[24];
    const float* cm_b    = (const float*)d_in[25];
    const float* ln2_g   = (const float*)d_in[26];
    const float* ln2_b   = (const float*)d_in[27];
    const float* head_W  = (const float*)d_in[28];
    const float* head_b  = (const float*)d_in[29];
    float* out = (float*)d_out;

    // workspace carving
    char* ws = (char*)d_ws;
    size_t off = 0;
    auto carve = [&](size_t bytes) -> void* {
        void* p = ws + off;
        off = (off + bytes + 255) & ~(size_t)255;
        return p;
    };
    const size_t SZF = (size_t)MTOK * Cc * sizeof(float);     // 134 MB
    float*  h_buf   = (float*)carve(SZF);
    float*  bufV    = (float*)carve(SZF);
    float*  bufR    = (float*)carve(SZF);
    __bf16* bufA    = (__bf16*)carve(SZF / 2);
    __bf16* bufRW   = (__bf16*)carve(SZF / 2);
    __bf16* bufT1   = (__bf16*)carve((size_t)MTOK * FF * 2);  // 268 MB
    float*  g1      = (float*)carve((size_t)Bc * 32 * Tc * sizeof(float));
    float*  stats   = (float*)carve((size_t)Bc * 2 * sizeof(float));
    __bf16* wv_t    = (__bf16*)carve((size_t)Cc * Cc * 2);
    __bf16* wr_t    = (__bf16*)carve((size_t)Cc * Cc * 2);
    __bf16* wo_t    = (__bf16*)carve((size_t)Cc * Cc * 2);
    __bf16* w1_t    = (__bf16*)carve((size_t)Cc * FF * 2);
    __bf16* w2_t    = (__bf16*)carve((size_t)FF * Cc * 2);
    (void)ws_size; (void)in_sizes; (void)n_in; (void)out_size;

    // ---- front end ----
    dec_kernel<<<MTOK, 256, 0, stream>>>(x, dec_Wt, dec_bt, dec_Wf, dec_bf, h_buf);
    conv1_kernel<<<(Bc * 32 * Tc) / 256, 256, 0, stream>>>(x, mg_W1, mg_b1, g1);
    conv2_kernel<<<(Bc * Cc * Tc) / 256, 256, 0, stream>>>(g1, mg_W2, mg_b2, bufV);
    bstats_kernel<<<Bc, 256, 0, stream>>>(bufV, stats);
    gadd_kernel<<<MTOK, 256, 0, stream>>>(bufV, stats, mg_g, mg_be, h_buf);

    const int gridCC = (MTOK / 128) * (Cc / 128);   // 2048
    const int gridCF = (MTOK / 128) * (FF / 128);   // 8192

    for (int l = 0; l < Lc; ++l) {
        // per-layer weight conversion (transpose + bf16)
        wconv_kernel<<<(Cc * Cc) / 256, 256, 0, stream>>>(tm_Wv + (size_t)l * Cc * Cc, wv_t, Cc, Cc);
        wconv_kernel<<<(Cc * Cc) / 256, 256, 0, stream>>>(tm_Wr + (size_t)l * Cc * Cc, wr_t, Cc, Cc);
        wconv_kernel<<<(Cc * Cc) / 256, 256, 0, stream>>>(tm_Wo + (size_t)l * Cc * Cc, wo_t, Cc, Cc);
        wconv_kernel<<<(Cc * FF) / 256, 256, 0, stream>>>(ffn_W1 + (size_t)l * Cc * FF, w1_t, Cc, FF);
        wconv_kernel<<<(FF * Cc) / 256, 256, 0, stream>>>(ffn_W2 + (size_t)l * FF * Cc, w2_t, FF, Cc);

        // hn = LN1(h) -> bf16
        ln_kernel<<<MTOK, 256, 0, stream>>>(h_buf, nullptr, ln1_g + l * Cc, ln1_b + l * Cc,
                                            nullptr, bufA);
        // v = hn @ Wv ; rpre = hn @ Wr
        gemm_wmma_kernel<G_STORE><<<gridCC, 256, 0, stream>>>(bufA, wv_t, nullptr, bufV, nullptr,
                                                              MTOK, Cc, Cc);
        gemm_wmma_kernel<G_STORE><<<gridCC, 256, 0, stream>>>(bufA, wr_t, nullptr, bufR, nullptr,
                                                              MTOK, Cc, Cc);
        // rw = sigmoid(rpre) * wkv(v)
        rwkv_scan_kernel<<<(Bc * Cc) / 256, 256, 0, stream>>>(bufV, bufR, tm_decay + l * Cc,
                                                              tm_first + l * Cc, bufRW);
        // h += rw @ Wo
        gemm_wmma_kernel<G_ADD><<<gridCC, 256, 0, stream>>>(bufRW, wo_t, nullptr, h_buf, nullptr,
                                                            MTOK, Cc, Cc);
        // hn2 = LN2(h) -> f32 (bufV) + bf16 (bufA)
        ln_kernel<<<MTOK, 256, 0, stream>>>(h_buf, nullptr, ln2_g + l * Cc, ln2_b + l * Cc,
                                            bufV, bufA);
        // t1 = gelu(hn2 @ W1 + b1) -> bf16
        gemm_wmma_kernel<G_GELU_BF16><<<gridCF, 256, 0, stream>>>(bufA, w1_t, ffn_b1 + l * FF,
                                                                  nullptr, bufT1, MTOK, FF, Cc);
        // ff = t1 @ W2 + b2 -> f32
        gemm_wmma_kernel<G_STORE><<<gridCC, 256, 0, stream>>>(bufT1, w2_t, ffn_b2 + l * Cc,
                                                              bufR, nullptr, MTOK, Cc, FF);
        // h = LN_cm(hn2 + ff)
        ln_kernel<<<MTOK, 256, 0, stream>>>(bufV, bufR, cm_g + l * Cc, cm_b + l * Cc,
                                            h_buf, nullptr);
    }

    head_kernel<<<Bc, 256, 0, stream>>>(h_buf, head_W, head_b, out);
}
